// DTTreeGRU_40596030882338
// MI455X (gfx1250) — matrix operand order repaired and checked by reading, not compiled, not run on hardware
//
#include <hip/hip_runtime.h>
#include <hip/hip_bf16.h>
#include <math.h>

// ---------------------------------------------------------------------------
// Tree-GRU on MI455X (gfx1250): bf16 WMMA GEMMs, async-to-LDS weight tiles,
// double-buffered LDS, level-parallel tree walk.
// L=128 nodes (binary heap), B=128, I=H=1024.
// ---------------------------------------------------------------------------

#define TL 128          // tree length (nodes)
#define TB 128          // batch
#define TI 1024         // input size
#define TH 1024         // hidden size
#define KG 3072         // K for both GEMMs: I + H + H
#define NG 5120         // gates N = 5*H
#define NC 1024         // cell  N = H
#define NKT (KG / 32)   // K-steps

#define APAD 40         // As row: 32 data + 8 pad ushorts (80 B, 16B-aligned)
#define BPAD 136        // Bs row: 128 data + 8 pad ushorts (272 B, 16B-aligned)

typedef __attribute__((ext_vector_type(16))) __bf16          v16bf;
typedef __attribute__((ext_vector_type(16))) unsigned short  v16u;
typedef __attribute__((ext_vector_type(8)))  float           v8f;

static __device__ __forceinline__ unsigned short f2bf_bits(float f) {
    unsigned u = __builtin_bit_cast(unsigned, f);
    unsigned r = u + 0x7fffu + ((u >> 16) & 1u);   // round-to-nearest-even
    return (unsigned short)(r >> 16);
}
static __device__ __forceinline__ float bf2f_bits(unsigned short s) {
    unsigned u = ((unsigned)s) << 16;
    return __builtin_bit_cast(float, u);
}

// Generic shared-pointer -> 32-bit LDS offset (low 32 bits of the aperture).
static __device__ __forceinline__ unsigned lds_off(const void* p) {
    return (unsigned)(unsigned long long)p;
}

// CDNA5 async global->LDS DMA (16 B per lane), tracked by ASYNCcnt.
static __device__ __forceinline__ void async_load_b128(unsigned lds_addr,
                                                       const void* gaddr) {
    asm volatile("global_load_async_to_lds_b128 %0, %1, off"
                 :: "v"(lds_addr), "v"(gaddr) : "memory");
}
static __device__ __forceinline__ void wait_asynccnt0() {
    asm volatile("s_wait_asynccnt 0x0" ::: "memory");
}

// A fragment: 16x32 bf16 per wave. 16-bit A layout (ISA 7.12.2):
// lane m (0..15): K=0..7 then K=16..23; lane m+16: K=8..15 then K=24..31.
static __device__ __forceinline__ v16bf load_a_frag(
        const unsigned short (*As)[APAD], int m, int lane) {
    int kb = (lane >> 4) * 8;
    v16u u;
#pragma unroll
    for (int t = 0; t < 8; ++t) {
        u[t]     = As[m][kb + t];
        u[8 + t] = As[m][kb + 16 + t];
    }
    return __builtin_bit_cast(v16bf, u);
}

// B fragment: 32x16 bf16 per wave. Lane l holds N=n0+l%16; K order matches A (K=8h..8h+7 then 16+8h..).
static __device__ __forceinline__ v16bf load_b_frag(
        const unsigned short (*Bs)[BPAD], int n0, int lane) {
    v16u u;
#pragma unroll
    for (int j = 0; j < 16; ++j) u[j] = Bs[(lane >> 4) * 8 + (j & 7) + ((j >> 3) << 4)][n0 + (lane & 15)];
    return __builtin_bit_cast(v16bf, u);
}

// ---------------------------------------------------------------------------
// Weight prep: stack + transpose + convert to bf16 in [K][N] layout.
// ---------------------------------------------------------------------------
__global__ __launch_bounds__(256) void prep_weights(
        const float* __restrict__ Wg_ih, const float* __restrict__ Wg_l,
        const float* __restrict__ Wg_r,  const float* __restrict__ Wc_ih,
        const float* __restrict__ Wc_l,  const float* __restrict__ Wc_r,
        unsigned short* __restrict__ Wg_bf, unsigned short* __restrict__ Wc_bf) {
    const size_t NGE = (size_t)KG * NG;
    const size_t NCE = (size_t)KG * NC;
    size_t i = (size_t)blockIdx.x * 256 + threadIdx.x;
    if (i < NGE) {
        int k = (int)(i / NG), n = (int)(i % NG);
        int reg = k >> 10, kk = k & 1023;
        const float* W = (reg == 0) ? Wg_ih : (reg == 1 ? Wg_l : Wg_r);
        Wg_bf[i] = f2bf_bits(W[(size_t)n * 1024 + kk]);
    } else if (i < NGE + NCE) {
        size_t j = i - NGE;
        int k = (int)(j / NC), n = (int)(j % NC);
        int reg = k >> 10, kk = k & 1023;
        const float* W = (reg == 0) ? Wc_ih : (reg == 1 ? Wc_l : Wc_r);
        Wc_bf[j] = f2bf_bits(W[(size_t)n * 1024 + kk]);
    }
}

__global__ __launch_bounds__(256) void zero_tail(float* __restrict__ out) {
    int i = blockIdx.x * 256 + threadIdx.x;
    if (i < TB * TH) out[(size_t)TB * TL * TH + i] = 0.0f;
}

// ---------------------------------------------------------------------------
// Gates GEMM: gates = sigmoid([x|lh|rh] @ Wg_stacked + bg), stored bf16.
// Block tile 64(M) x 128(N), K-step 32; 8 waves, each 16x64 (4 WMMAs).
// Double-buffered LDS; B tiles via async global->LDS DMA.
// ---------------------------------------------------------------------------
__global__ __launch_bounds__(256) void gates_gemm(
        const float* __restrict__ inputs, const float* __restrict__ Hst,
        const unsigned short* __restrict__ Wg_bf, const float* __restrict__ bg,
        const int* __restrict__ lchild, const int* __restrict__ rchild,
        unsigned short* __restrict__ gates, const float* __restrict__ zpage,
        int node_base) {
    __shared__ unsigned short As[2][64][APAD];
    __shared__ unsigned short Bs[2][32][BPAD];

    const int tid  = threadIdx.x;
    const int lane = tid & 31;
    const int wave = tid >> 5;
    const int wm   = wave & 3;        // wave row (0..3) -> 16 rows each
    const int wn   = wave >> 2;       // wave col (0..1) -> 64 cols each
    const int ln   = lane & 15;
    const int lh2  = lane >> 4;
    const int ncol0 = blockIdx.x * 128;

    // per-thread A-stage metadata (one row, 8 contiguous K per step)
    const int arow = tid >> 2;                  // 0..63
    const int k0   = (tid & 3) << 3;            // 0,8,16,24
    const int rg   = blockIdx.y * 64 + arow;    // chunk-local row
    const int node = node_base + (rg >> 7);
    const int b    = rg & 127;
    const float* xrow = inputs + ((size_t)(node * TB + b)) * TI;
    const int lidx = lchild[b * TL + node];
    const int ridx = rchild[b * TL + node];
    // absent children read the permanently-zero h_n tail of d_out (zpage)
    const float* lrow = (lidx >= 0) ? Hst + ((size_t)(b * TL + lidx)) * TH : zpage;
    const float* rrow = (ridx >= 0) ? Hst + ((size_t)(b * TL + ridx)) * TH : zpage;

    // per-thread B-stage metadata (two async b128 per k-step)
    const int bkr = tid >> 3;                   // 0..31
    const int bn0 = (tid & 7) << 4;             // 0..112

    v8f acc[4];
#pragma unroll
    for (int s = 0; s < 4; ++s) acc[s] = (v8f){0,0,0,0,0,0,0,0};

    // prologue: kt=0 is always the x region
    {
#pragma unroll
        for (int j = 0; j < 8; ++j) As[0][arow][k0 + j] = f2bf_bits(xrow[k0 + j]);
        const unsigned short* g = Wg_bf + (size_t)bkr * NG + ncol0 + bn0;
        unsigned d = lds_off(&Bs[0][bkr][bn0]);
        async_load_b128(d, g);
        async_load_b128(d + 16, g + 8);
    }

    for (int kt = 0; kt < NKT; ++kt) {
        const int cur = kt & 1;
        wait_asynccnt0();          // B[cur] resident in LDS
        __syncthreads();           // A[cur] stores visible; prev reads done
        if (kt + 1 < NKT) {        // overlap next tile with this tile's WMMAs
            const int nxt = cur ^ 1;
            const int kt2 = kt + 1;
            const unsigned short* g =
                Wg_bf + (size_t)(kt2 * 32 + bkr) * NG + ncol0 + bn0;
            unsigned d = lds_off(&Bs[nxt][bkr][bn0]);
            async_load_b128(d, g);
            async_load_b128(d + 16, g + 8);
            const int reg = kt2 >> 5;            // block-uniform region select
            const int kk  = ((kt2 & 31) << 5) + k0;
            const float* src = (reg == 0) ? xrow : ((reg == 1) ? lrow : rrow);
#pragma unroll
            for (int j = 0; j < 8; ++j)
                As[nxt][arow][k0 + j] = f2bf_bits(src[kk + j]);
        }
        v16bf af = load_a_frag(As[cur], wm * 16 + ln, lane);
#pragma unroll
        for (int s = 0; s < 4; ++s) {
            v16bf bf = load_b_frag(Bs[cur], wn * 64 + s * 16, lane);
            acc[s] = __builtin_amdgcn_wmma_f32_16x16x32_bf16(
                false, af, false, bf, (short)0, acc[s], false, false);
        }
    }

    // epilogue: +bias, sigmoid, store bf16 gates
#pragma unroll
    for (int s = 0; s < 4; ++s) {
#pragma unroll
        for (int v = 0; v < 8; ++v) {
            int n = ncol0 + wn * 64 + s * 16 + ln;
            int m = blockIdx.y * 64 + wm * 16 + v + 8 * lh2;
            float val = acc[s][v] + bg[n];
            float g = 1.0f / (1.0f + __expf(-val));
            gates[(size_t)m * NG + n] = f2bf_bits(g);
        }
    }
}

// ---------------------------------------------------------------------------
// Cell GEMM + fused update:
//   cell = tanh([x | rl*lh | rr*rh] @ Wc_stacked + bc)
//   h    = zl*lh + zr*rh + z*cell   -> written into d_out H-state
// ---------------------------------------------------------------------------
__global__ __launch_bounds__(256) void cell_gemm(
        const float* __restrict__ inputs, float* __restrict__ Hst,
        const unsigned short* __restrict__ Wc_bf, const float* __restrict__ bc,
        const int* __restrict__ lchild, const int* __restrict__ rchild,
        const unsigned short* __restrict__ gates, const float* __restrict__ zpage,
        int node_base) {
    __shared__ unsigned short As[2][64][APAD];
    __shared__ unsigned short Bs[2][32][BPAD];

    const int tid  = threadIdx.x;
    const int lane = tid & 31;
    const int wave = tid >> 5;
    const int wm   = wave & 3;
    const int wn   = wave >> 2;
    const int ln   = lane & 15;
    const int lh2  = lane >> 4;
    const int ncol0 = blockIdx.x * 128;

    const int arow = tid >> 2;
    const int k0   = (tid & 3) << 3;
    const int rg   = blockIdx.y * 64 + arow;
    const int node = node_base + (rg >> 7);
    const int b    = rg & 127;
    const float* xrow = inputs + ((size_t)(node * TB + b)) * TI;
    const int lidx = lchild[b * TL + node];
    const int ridx = rchild[b * TL + node];
    const float* lrow = (lidx >= 0) ? Hst + ((size_t)(b * TL + lidx)) * TH : zpage;
    const float* rrow = (ridx >= 0) ? Hst + ((size_t)(b * TL + ridx)) * TH : zpage;
    const unsigned short* grow = gates + (size_t)rg * NG;

    const int bkr = tid >> 3;
    const int bn0 = (tid & 7) << 4;

    v8f acc[4];
#pragma unroll
    for (int s = 0; s < 4; ++s) acc[s] = (v8f){0,0,0,0,0,0,0,0};

    // prologue: kt=0 is always the x region
    {
#pragma unroll
        for (int j = 0; j < 8; ++j) As[0][arow][k0 + j] = f2bf_bits(xrow[k0 + j]);
        const unsigned short* g = Wc_bf + (size_t)bkr * NC + ncol0 + bn0;
        unsigned d = lds_off(&Bs[0][bkr][bn0]);
        async_load_b128(d, g);
        async_load_b128(d + 16, g + 8);
    }

    for (int kt = 0; kt < NKT; ++kt) {
        const int cur = kt & 1;
        wait_asynccnt0();
        __syncthreads();
        if (kt + 1 < NKT) {
            const int nxt = cur ^ 1;
            const int kt2 = kt + 1;
            const unsigned short* g =
                Wc_bf + (size_t)(kt2 * 32 + bkr) * NC + ncol0 + bn0;
            unsigned d = lds_off(&Bs[nxt][bkr][bn0]);
            async_load_b128(d, g);
            async_load_b128(d + 16, g + 8);
            const int reg = kt2 >> 5;            // block-uniform region select
            const int kk  = ((kt2 & 31) << 5) + k0;
            if (reg == 0) {
#pragma unroll
                for (int j = 0; j < 8; ++j)
                    As[nxt][arow][k0 + j] = f2bf_bits(xrow[kk + j]);
            } else {
                const float* hsrc = (reg == 1) ? lrow : rrow;
                const unsigned short* gsrc =
                    grow + ((reg == 1) ? 0 : TH) + kk;   // rl or rr gate slice
#pragma unroll
                for (int j = 0; j < 8; ++j) {
                    float rv = bf2f_bits(gsrc[j]);
                    As[nxt][arow][k0 + j] = f2bf_bits(rv * hsrc[kk + j]);
                }
            }
        }
        v16bf af = load_a_frag(As[cur], wm * 16 + ln, lane);
#pragma unroll
        for (int s = 0; s < 4; ++s) {
            v16bf bf = load_b_frag(Bs[cur], wn * 64 + s * 16, lane);
            acc[s] = __builtin_amdgcn_wmma_f32_16x16x32_bf16(
                false, af, false, bf, (short)0, acc[s], false, false);
        }
    }

    // epilogue: tanh + tree-GRU combine, write h into d_out H-state
#pragma unroll
    for (int s = 0; s < 4; ++s) {
#pragma unroll
        for (int v = 0; v < 8; ++v) {
            int h = ncol0 + wn * 64 + s * 16 + ln;                   // 0..1023
            int m = blockIdx.y * 64 + wm * 16 + v + 8 * lh2;
            int nd = node_base + (m >> 7);
            int bb = m & 127;
            int li = lchild[bb * TL + nd];
            int ri = rchild[bb * TL + nd];
            float lhv = (li >= 0) ? Hst[((size_t)(bb * TL + li)) * TH + h] : 0.0f;
            float rhv = (ri >= 0) ? Hst[((size_t)(bb * TL + ri)) * TH + h] : 0.0f;
            const unsigned short* gr = gates + (size_t)m * NG;
            float zl = bf2f_bits(gr[2 * TH + h]);
            float zr = bf2f_bits(gr[3 * TH + h]);
            float zz = bf2f_bits(gr[4 * TH + h]);
            float cell = tanhf(acc[s][v] + bc[h]);
            Hst[((size_t)(bb * TL + nd)) * TH + h] = zl * lhv + zr * rhv + zz * cell;
        }
    }
}

// ---------------------------------------------------------------------------
// Host launcher
// ---------------------------------------------------------------------------
extern "C" void kernel_launch(void* const* d_in, const int* in_sizes, int n_in,
                              void* d_out, int out_size, void* d_ws, size_t ws_size,
                              hipStream_t stream) {
    const float* inputs = (const float*)d_in[0];
    const float* Wg_ih  = (const float*)d_in[1];
    const float* bg_ih  = (const float*)d_in[2];
    const float* Wg_l   = (const float*)d_in[3];
    const float* Wg_r   = (const float*)d_in[4];
    const float* Wc_ih  = (const float*)d_in[5];
    const float* bc_ih  = (const float*)d_in[6];
    const float* Wc_l   = (const float*)d_in[7];
    const float* Wc_r   = (const float*)d_in[8];
    // d_in[9] = indexes (fixed heap reversal; order derived on host)
    const int* lchild = (const int*)d_in[10];
    const int* rchild = (const int*)d_in[11];

    float* Hst = (float*)d_out;   // Hfin [B, L, H] lives directly in d_out
    const float* zpage = Hst + (size_t)TB * TL * TH;  // zeroed h_n tail

    // workspace layout
    unsigned short* Wg_bf = (unsigned short*)d_ws;                       // 3072*5120
    unsigned short* Wc_bf = Wg_bf + (size_t)KG * NG;                     // 3072*1024
    unsigned short* gatbf = Wc_bf + (size_t)KG * NC;                     // 2048*5120 max

    // 1) one-shot weight conversion + output tail zeroing (zpage source)
    {
        size_t tot = (size_t)KG * NG + (size_t)KG * NC;
        int blocks = (int)((tot + 255) / 256);
        prep_weights<<<blocks, 256, 0, stream>>>(Wg_ih, Wg_l, Wg_r,
                                                 Wc_ih, Wc_l, Wc_r, Wg_bf, Wc_bf);
        zero_tail<<<(TB * TH + 255) / 256, 256, 0, stream>>>(Hst);
    }

    // 2) level-parallel tree walk, deepest level first (valid topo order).
    for (int lev = 7; lev >= 0; --lev) {
        int lo = (1 << lev) - 1;
        if (lo >= TL) continue;
        int hi = (1 << (lev + 1)) - 2;
        if (hi >= TL) hi = TL - 1;
        int cnt = hi - lo + 1;
        for (int s = 0; s < cnt; s += 16) {          // chunk <= 16 nodes
            int c = (cnt - s < 16) ? (cnt - s) : 16;
            int M = c * TB;                          // rows this chunk
            int nbase = lo + s;
            dim3 gg(NG / 128, M / 64);
            gates_gemm<<<gg, 256, 0, stream>>>(inputs, Hst, Wg_bf, bg_ih,
                                               lchild, rchild, gatbf, zpage, nbase);
            dim3 gc(NC / 128, M / 64);
            cell_gemm<<<gc, 256, 0, stream>>>(inputs, Hst, Wc_bf, bc_ih,
                                              lchild, rchild, gatbf, zpage, nbase);
        }
    }
}
